// MambaLLM_25546465477284
// MI455X (gfx1250) — matrix operand order repaired and checked by reading, not compiled, not run on hardware
//
#include <hip/hip_runtime.h>
#include <math.h>

// ---------------- model constants ----------------
#define LSEQ 2048
#define DIM  1024
#define NLAY 12
#define VOCAB 50257
#define NST  16
#define KCONV 4
#define EDIM 2048
#define DTR  64
#define EPSR 1e-5f
#define TCHUNK 64   // timesteps of dbc staged per TDM transfer

// ---------------- TDM availability (this toolchain: 6-arg builtin) ----------
#if defined(__has_builtin)
# if __has_builtin(__builtin_amdgcn_tensor_load_to_lds)
#  define HAVE_TDM 1
# endif
#endif
#ifndef HAVE_TDM
# define HAVE_TDM 0
#endif

// ---------------- vector types ----------------
typedef __attribute__((ext_vector_type(16))) __bf16    v16bf;
typedef __attribute__((ext_vector_type(8)))  float     v8f;
typedef __attribute__((ext_vector_type(4)))  float     f4;
typedef __attribute__((ext_vector_type(4)))  unsigned  u32x4;
typedef __attribute__((ext_vector_type(8)))  int       i32x8;
typedef __attribute__((ext_vector_type(4)))  int       i32x4;

// pack two f32 into one dword of 2x bf16 (truncating) with a single v_perm_b32
__device__ __forceinline__ unsigned pk_bf16(float lo, float hi) {
  unsigned a, b;
  __builtin_memcpy(&a, &lo, 4);
  __builtin_memcpy(&b, &hi, 4);
  return __builtin_amdgcn_perm(b, a, 0x07060302u);
}

// A fragment, 16x32 bf16: this lane's elements are ap[0..15] (contiguous K)
__device__ __forceinline__ v16bf load_fragA(const float* __restrict__ ap) {
  const f4* p = (const f4*)ap;
  f4 x0 = p[0], x1 = p[1], x2 = p[2], x3 = p[3];
  union { unsigned u[8]; v16bf v; } r;
  r.u[0] = pk_bf16(x0[0], x0[1]);
  r.u[1] = pk_bf16(x0[2], x0[3]);
  r.u[2] = pk_bf16(x1[0], x1[1]);
  r.u[3] = pk_bf16(x1[2], x1[3]);
  r.u[4] = pk_bf16(x2[0], x2[1]);
  r.u[5] = pk_bf16(x2[2], x2[3]);
  r.u[6] = pk_bf16(x3[0], x3[1]);
  r.u[7] = pk_bf16(x3[2], x3[3]);
  return r.v;
}

// B fragment, 32x16 bf16: this lane's elements are bp[0..15] (contiguous K)
__device__ __forceinline__ v16bf load_fragB_full(const float* __restrict__ bp) {
  const f4* p = (const f4*)bp;
  f4 x0 = p[0], x1 = p[1], x2 = p[2], x3 = p[3];
  union { unsigned u[8]; v16bf v; } r;
  r.u[0] = pk_bf16(x0[0], x0[1]);
  r.u[1] = pk_bf16(x0[2], x0[3]);
  r.u[2] = pk_bf16(x1[0], x1[1]);
  r.u[3] = pk_bf16(x1[2], x1[3]);
  r.u[4] = pk_bf16(x2[0], x2[1]);
  r.u[5] = pk_bf16(x2[2], x2[3]);
  r.u[6] = pk_bf16(x3[0], x3[1]);
  r.u[7] = pk_bf16(x3[2], x3[3]);
  return r.v;
}

// guarded variant: `full` is wave-uniform; lanevalid zeroes OOB columns via cndmask
__device__ __forceinline__ v16bf load_fragB(const float* __restrict__ bp,
                                            bool full, bool lanevalid) {
  const f4* p = (const f4*)bp;
  f4 x0 = p[0], x1 = p[1], x2 = p[2], x3 = p[3];
  union { unsigned u[8]; v16bf v; } r;
  r.u[0] = pk_bf16(x0[0], x0[1]);
  r.u[1] = pk_bf16(x0[2], x0[3]);
  r.u[2] = pk_bf16(x1[0], x1[1]);
  r.u[3] = pk_bf16(x1[2], x1[3]);
  r.u[4] = pk_bf16(x2[0], x2[1]);
  r.u[5] = pk_bf16(x2[2], x2[3]);
  r.u[6] = pk_bf16(x3[0], x3[1]);
  r.u[7] = pk_bf16(x3[2], x3[3]);
  if (!full) {
#pragma unroll
    for (int i = 0; i < 8; ++i) r.u[i] = lanevalid ? r.u[i] : 0u;
  }
  return r.v;
}

// ---------------- embedding gather ----------------
__global__ void embed_kernel(const int* __restrict__ ids,
                             const float* __restrict__ emb,
                             float* __restrict__ x) {
  int idx = blockIdx.x * blockDim.x + threadIdx.x;
  if (idx >= LSEQ * DIM) return;
  int t = idx / DIM, d = idx % DIM;
  x[idx] = emb[(size_t)ids[t] * DIM + d];
}

// ---------------- residual add + RMSNorm (one block per row) ----------------
__global__ void add_rmsnorm_kernel(const float* __restrict__ xin,
                                   float* __restrict__ res,
                                   float* __restrict__ h,
                                   const float* __restrict__ w,
                                   int addRes) {
  int row = blockIdx.x;
  const float* xr = xin + (size_t)row * DIM;
  float* rr = res + (size_t)row * DIM;
  float vals[4];
  float local = 0.f;
#pragma unroll
  for (int i = 0; i < 4; ++i) {
    int d = threadIdx.x + i * 256;
    float v = xr[d];
    if (addRes) v += rr[d];
    rr[d] = v;
    vals[i] = v;
    local += v * v;
  }
  __shared__ float red[256];
  red[threadIdx.x] = local;
  __syncthreads();
  for (int s = 128; s > 0; s >>= 1) {
    if (threadIdx.x < s) red[threadIdx.x] += red[threadIdx.x + s];
    __syncthreads();
  }
  float inv = rsqrtf(red[0] / (float)DIM + EPSR);
#pragma unroll
  for (int i = 0; i < 4; ++i) {
    int d = threadIdx.x + i * 256;
    h[(size_t)row * DIM + d] = vals[i] * inv * w[d];
  }
}

// ---------------- causal depthwise conv (K=4) + SiLU ----------------
__global__ void conv_silu_kernel(const float* __restrict__ xz,
                                 const float* __restrict__ cw,
                                 const float* __restrict__ cb,
                                 float* __restrict__ xc) {
  int idx = blockIdx.x * blockDim.x + threadIdx.x;
  if (idx >= LSEQ * EDIM) return;
  int e = idx % EDIM, t = idx / EDIM;
  float acc = cb[e];
#pragma unroll
  for (int j = 0; j < KCONV; ++j) {
    int tt = t - (KCONV - 1) + j;
    if (tt >= 0) acc += cw[e * KCONV + j] * xz[(size_t)tt * (2 * EDIM) + e];
  }
  xc[idx] = acc / (1.f + __expf(-acc));
}

// ---------------- TDM: stage a TCHUNKx96 f32 tile of dbc into LDS ----------
#if HAVE_TDM
__device__ __forceinline__ void tdm_load_dbc(float* lds_dst,
                                             const float* __restrict__ dbc,
                                             int row0) {
  unsigned lds_off = (unsigned)(uintptr_t)lds_dst;   // low 32 bits = LDS offset
  unsigned long long ga = (unsigned long long)(uintptr_t)(dbc + (size_t)row0 * 96);
  unsigned td0 = 96u;                       // tensor dim0 (elements per row)
  unsigned td1 = (unsigned)(LSEQ - row0);   // remaining rows from tile start
  u32x4 g0 = { 1u,                          // count=1, user mode
               lds_off,                     // lds_addr
               (unsigned)ga,                // global_addr[31:0]
               (unsigned)((ga >> 32) & 0x01FFFFFFu) | (2u << 30) }; // [56:32] | type=2
  i32x8 g1;
  g1[0] = (int)(2u << 16);                                          // data_size = 4B
  g1[1] = (int)((td0 & 0xFFFFu) << 16);                             // dim0[15:0]
  g1[2] = (int)(((td0 >> 16) & 0xFFFFu) | ((td1 & 0xFFFFu) << 16)); // dim0 hi | dim1 lo
  g1[3] = (int)(((td1 >> 16) & 0xFFFFu) | (96u << 16));             // dim1 hi | tile_dim0=96
  g1[4] = (int)(TCHUNK & 0xFFFF);                                   // tile_dim1, tile_dim2=0
  g1[5] = (int)96;                                                  // tensor_dim0_stride
  g1[6] = 0;
  g1[7] = 0;
  i32x4 gz4 = { 0, 0, 0, 0 };
  i32x8 gz8 = { 0, 0, 0, 0, 0, 0, 0, 0 };
  // 6-arg form: (g0, g1, g2, g3, extra_group, cpol)
  __builtin_amdgcn_tensor_load_to_lds(g0, g1, gz4, gz4, gz8, 0);
}
#endif

// ---------------- selective scan (TDM double-buffered B/C staging) ----------
__global__ void scan_kernel(const float* __restrict__ dtb,   // [L,E] softplus(dt)
                            const float* __restrict__ dbc,   // [L,96] (dt|B|C)
                            const float* __restrict__ xc,    // [L,E]
                            const float* __restrict__ xz,    // [L,2E] (z = cols E..2E)
                            const float* __restrict__ A_log, // [E,N]
                            const float* __restrict__ Dp,    // [E]
                            float* __restrict__ y) {         // [L,E]
#if HAVE_TDM
  __shared__ float sdbc[2][TCHUNK * 96];   // 2 x 24KB double buffer
#endif
  int gid = blockIdx.x * blockDim.x + threadIdx.x;  // grid is exact: E*16 lanes
  int e = gid >> 4;
  int n = gid & 15;
  float A = -__expf(A_log[e * NST + n]);
  float Dv = Dp[e];
  float hst = 0.f;

#if HAVE_TDM
  if (threadIdx.x < 32) {
    tdm_load_dbc(&sdbc[0][0], dbc, 0);
    __builtin_amdgcn_s_wait_tensorcnt(0);
  }
  __syncthreads();
#endif

  for (int c = 0; c < LSEQ / TCHUNK; ++c) {
#if HAVE_TDM
    if ((c + 1) < LSEQ / TCHUNK && threadIdx.x < 32)
      tdm_load_dbc(&sdbc[(c + 1) & 1][0], dbc, (c + 1) * TCHUNK);
    const float* bc = &sdbc[c & 1][0];
#else
    const float* bc = dbc + (size_t)c * TCHUNK * 96;
#endif
    for (int tt = 0; tt < TCHUNK; ++tt) {
      int t = c * TCHUNK + tt;
      float dt  = dtb[(size_t)t * EDIM + e];
      float xcv = xc[(size_t)t * EDIM + e];
      float Bn  = bc[tt * 96 + DTR + n];
      float Cn  = bc[tt * 96 + DTR + NST + n];
      float dA  = __expf(dt * A);
      hst = dA * hst + dt * Bn * xcv;
      float p = hst * Cn;
      p += __shfl_xor(p, 1);
      p += __shfl_xor(p, 2);
      p += __shfl_xor(p, 4);
      p += __shfl_xor(p, 8);
      if (n == 0) {
        float zv = xz[(size_t)t * (2 * EDIM) + EDIM + e];
        float yv = p + Dv * xcv;
        yv *= zv / (1.f + __expf(-zv));   // * silu(z)
        y[(size_t)t * EDIM + e] = yv;
      }
    }
#if HAVE_TDM
    if (threadIdx.x < 32) __builtin_amdgcn_s_wait_tensorcnt(0);
    __syncthreads();
#endif
  }
}

// ---------------- epilogue store for one 16x16 tile column ----------------
__device__ __forceinline__ void store_tile(float* __restrict__ C, int ldc,
                                           const float* __restrict__ bias, int act,
                                           int mrow, int n, int N, v8f acc) {
  if (n >= N) return;
  float bv = bias ? bias[n] : 0.f;
#pragma unroll
  for (int r = 0; r < 8; ++r) {
    float v = acc[r] + bv;
    if (act) v = (v > 20.f) ? v : log1pf(__expf(v));  // softplus
    C[(size_t)(mrow + r) * ldc + n] = v;
  }
}

// ---------------- WMMA GEMM:  C[M,N] = A[M,K] * W[N,K]^T (+bias, +softplus) ----
// One wave computes a 32x64 strip: 2 A fragments x 4 shared B fragments ->
// 8 WMMAs per K-step. All six fragments are materialized (in distinct
// registers) BEFORE any WMMA so the 24 b128 loads can clause/overlap and no
// WMMA->VALU WAR hazards (v_nop) are needed inside the step.
__global__ void gemm_nt_wmma(const float* __restrict__ A,
                             const float* __restrict__ W,
                             const float* __restrict__ bias,
                             float* __restrict__ C,
                             int M, int N, int K,
                             int lda, int ldw, int ldc, int act) {
  int lane = threadIdx.x & 31;
  int tile = __builtin_amdgcn_readfirstlane(
      (int)(blockIdx.x * (blockDim.x >> 5) + (threadIdx.x >> 5)));
  int nstrips = (N + 63) >> 6;
  int total = (M >> 5) * nstrips;
  if (tile >= total) return;
  int mt = tile / nstrips;
  int ns = tile - mt * nstrips;
  int m0 = mt << 5;
  int n0 = ns << 6;
  int half = lane >> 4;   // 0: lanes 0-15, 1: lanes 16-31
  int l15  = lane & 15;

  const float* arow0 = A + (size_t)(m0 + l15) * lda + half * 16;
  const float* arow1 = arow0 + (size_t)16 * lda;

  v8f acc00 = {0,0,0,0,0,0,0,0}, acc01 = {0,0,0,0,0,0,0,0};
  v8f acc02 = {0,0,0,0,0,0,0,0}, acc03 = {0,0,0,0,0,0,0,0};
  v8f acc10 = {0,0,0,0,0,0,0,0}, acc11 = {0,0,0,0,0,0,0,0};
  v8f acc12 = {0,0,0,0,0,0,0,0}, acc13 = {0,0,0,0,0,0,0,0};

  if (n0 + 64 <= N) {
    // ---------------- fast path: strip fully in-bounds, zero branches -------
    const float* b0 = W + (size_t)(n0 + l15) * ldw + half * 16;
    const float* b1 = b0 + (size_t)16 * ldw;
    const float* b2 = b1 + (size_t)16 * ldw;
    const float* b3 = b2 + (size_t)16 * ldw;
    for (int k0 = 0; k0 < K; k0 += 32) {
      __builtin_prefetch(arow0 + k0 + 32, 0, 3);
      __builtin_prefetch(b0 + k0 + 32, 0, 3);
      // materialize ALL fragments first (independent loads -> clauses)
      v16bf a0 = load_fragA(arow0 + k0);
      v16bf a1 = load_fragA(arow1 + k0);
      v16bf f0 = load_fragB_full(b0 + k0);
      v16bf f1 = load_fragB_full(b1 + k0);
      v16bf f2 = load_fragB_full(b2 + k0);
      v16bf f3 = load_fragB_full(b3 + k0);
      // then 8 back-to-back WMMAs
      acc00 = __builtin_amdgcn_wmma_f32_16x16x32_bf16(false, a0, false, f0, (short)0, acc00, false, false);
      acc10 = __builtin_amdgcn_wmma_f32_16x16x32_bf16(false, a1, false, f0, (short)0, acc10, false, false);
      acc01 = __builtin_amdgcn_wmma_f32_16x16x32_bf16(false, a0, false, f1, (short)0, acc01, false, false);
      acc11 = __builtin_amdgcn_wmma_f32_16x16x32_bf16(false, a1, false, f1, (short)0, acc11, false, false);
      acc02 = __builtin_amdgcn_wmma_f32_16x16x32_bf16(false, a0, false, f2, (short)0, acc02, false, false);
      acc12 = __builtin_amdgcn_wmma_f32_16x16x32_bf16(false, a1, false, f2, (short)0, acc12, false, false);
      acc03 = __builtin_amdgcn_wmma_f32_16x16x32_bf16(false, a0, false, f3, (short)0, acc03, false, false);
      acc13 = __builtin_amdgcn_wmma_f32_16x16x32_bf16(false, a1, false, f3, (short)0, acc13, false, false);
    }
  } else {
    // ---------------- guarded tail path (flags are scalar/uniform) ----------
    const float* brow[4];
    bool lvalid[4];
#pragma unroll
    for (int j = 0; j < 4; ++j) {
      int nn = n0 + j * 16 + l15;
      lvalid[j] = nn < N;
      brow[j] = W + (size_t)(lvalid[j] ? nn : (N - 1)) * ldw + half * 16;
    }
    for (int k0 = 0; k0 < K; k0 += 32) {
      v16bf a0 = load_fragA(arow0 + k0);
      v16bf a1 = load_fragA(arow1 + k0);
#define GEMM_TILE(J, ACC0, ACC1)                                               \
      if (n0 + (J) * 16 < N) {                                                 \
        v16bf b = load_fragB(brow[J] + k0, (n0 + (J) * 16 + 16) <= N, lvalid[J]); \
        ACC0 = __builtin_amdgcn_wmma_f32_16x16x32_bf16(                        \
            false, a0, false, b, (short)0, ACC0, false, false);                \
        ACC1 = __builtin_amdgcn_wmma_f32_16x16x32_bf16(                        \
            false, a1, false, b, (short)0, ACC1, false, false);                \
      }
      GEMM_TILE(0, acc00, acc10)
      GEMM_TILE(1, acc01, acc11)
      GEMM_TILE(2, acc02, acc12)
      GEMM_TILE(3, acc03, acc13)
#undef GEMM_TILE
    }
  }

  int mr0 = m0 + half * 8;
  int mr1 = m0 + 16 + half * 8;
  store_tile(C, ldc, bias, act, mr0, n0 +  0 + l15, N, acc00);
  store_tile(C, ldc, bias, act, mr1, n0 +  0 + l15, N, acc10);
  store_tile(C, ldc, bias, act, mr0, n0 + 16 + l15, N, acc01);
  store_tile(C, ldc, bias, act, mr1, n0 + 16 + l15, N, acc11);
  store_tile(C, ldc, bias, act, mr0, n0 + 32 + l15, N, acc02);
  store_tile(C, ldc, bias, act, mr1, n0 + 32 + l15, N, acc12);
  store_tile(C, ldc, bias, act, mr0, n0 + 48 + l15, N, acc03);
  store_tile(C, ldc, bias, act, mr1, n0 + 48 + l15, N, acc13);
}

// ---------------- host-side orchestration ----------------
static inline void launch_gemm(const float* A, const float* W, const float* bias,
                               float* C, int M, int N, int K,
                               int lda, int ldw, int ldc, int act,
                               hipStream_t stream) {
  int nstrips = (N + 63) >> 6;
  int total = (M >> 5) * nstrips;        // 32x64 wave tiles
  int blocks = (total + 7) / 8;          // 8 waves / 256-thread block (wave32)
  gemm_nt_wmma<<<blocks, 256, 0, stream>>>(A, W, bias, C, M, N, K, lda, ldw, ldc, act);
}

extern "C" void kernel_launch(void* const* d_in, const int* in_sizes, int n_in,
                              void* d_out, int out_size, void* d_ws, size_t ws_size,
                              hipStream_t stream) {
  (void)in_sizes; (void)n_in; (void)out_size; (void)ws_size;
  const int*   ids       = (const int*)  d_in[0];
  const float* embed     = (const float*)d_in[1];
  const float* in_proj_w = (const float*)d_in[2];
  const float* conv_w    = (const float*)d_in[3];
  const float* conv_b    = (const float*)d_in[4];
  const float* x_proj_w  = (const float*)d_in[5];
  const float* dt_proj_w = (const float*)d_in[6];
  const float* dt_proj_b = (const float*)d_in[7];
  const float* A_log     = (const float*)d_in[8];
  const float* D_param   = (const float*)d_in[9];
  const float* out_proj_w= (const float*)d_in[10];
  const float* norm_w    = (const float*)d_in[11];
  const float* norm_f_w  = (const float*)d_in[12];
  const float* head_w    = (const float*)d_in[13];
  float* logits = (float*)d_out;

  // workspace layout (float counts)
  const size_t LD = (size_t)LSEQ * DIM;     // 2M
  const size_t LE = (size_t)LSEQ * EDIM;    // 4M
  float* ws   = (float*)d_ws;
  float* res  = ws;                 // LD
  float* x    = res  + LD;          // LD
  float* h    = x    + LD;          // LD
  float* xz   = h    + LD;          // 2*LE
  float* xc   = xz   + 2 * LE;      // LE
  float* dbc  = xc   + LE;          // L*96
  float* dtb  = dbc  + (size_t)LSEQ * 96; // LE
  float* ybuf = dtb  + LE;          // LE

  // 1) embedding
  embed_kernel<<<(LSEQ * DIM + 255) / 256, 256, 0, stream>>>(ids, embed, x);

  // 2) layers
  for (int i = 0; i < NLAY; ++i) {
    add_rmsnorm_kernel<<<LSEQ, 256, 0, stream>>>(x, res, h, norm_w + (size_t)i * DIM, i > 0 ? 1 : 0);

    // xz = h @ in_proj_w[i]^T   (M=2048, N=4096, K=1024)
    launch_gemm(h, in_proj_w + (size_t)i * 2 * EDIM * DIM, nullptr, xz,
                LSEQ, 2 * EDIM, DIM, DIM, DIM, 2 * EDIM, 0, stream);

    // xc = silu(causal_dw_conv(xi))
    conv_silu_kernel<<<(LSEQ * EDIM + 255) / 256, 256, 0, stream>>>(
        xz, conv_w + (size_t)i * EDIM * KCONV, conv_b + (size_t)i * EDIM, xc);

    // dbc = xc @ x_proj_w[i]^T  (M=2048, N=96, K=2048)
    launch_gemm(xc, x_proj_w + (size_t)i * 96 * EDIM, nullptr, dbc,
                LSEQ, 96, EDIM, EDIM, EDIM, 96, 0, stream);

    // dt = softplus(dbc[:, :64] @ dt_proj_w[i]^T + dt_b)  (M=2048, N=2048, K=64)
    launch_gemm(dbc, dt_proj_w + (size_t)i * EDIM * DTR, dt_proj_b + (size_t)i * EDIM, dtb,
                LSEQ, EDIM, DTR, 96, DTR, EDIM, 1, stream);

    // selective scan -> ybuf  (E*16 lanes; TDM-staged B/C)
    scan_kernel<<<(EDIM * NST) / 256, 256, 0, stream>>>(
        dtb, dbc, xc, xz,
        A_log + (size_t)i * EDIM * NST, D_param + (size_t)i * EDIM, ybuf);

    // x = ybuf @ out_proj_w[i]^T  (M=2048, N=1024, K=2048)
    launch_gemm(ybuf, out_proj_w + (size_t)i * DIM * EDIM, nullptr, x,
                LSEQ, DIM, EDIM, EDIM, EDIM, DIM, 0, stream);
  }

  // 3) final norm + LM head
  add_rmsnorm_kernel<<<LSEQ, 256, 0, stream>>>(x, res, h, norm_f_w, 1);
  launch_gemm(h, head_w, nullptr, logits, LSEQ, VOCAB, DIM, DIM, DIM, VOCAB, 0, stream);
}
